// GCRNNCell_19078244728957
// MI455X (gfx1250) — matrix operand order repaired and checked by reading, not compiled, not run on hardware
//
#include <hip/hip_runtime.h>

#define B_ 64
#define N_ 1024
#define C_ 256
#define G_ 768
#define E_ 16384
#define M_ (B_ * N_)   // 65536 rows

typedef __attribute__((ext_vector_type(16))) __bf16 v16bf;
typedef __attribute__((ext_vector_type(8)))  float  v8f;

union FragBF {
    v16bf v;
    unsigned short u[16];
};

__device__ __forceinline__ unsigned short f2bf(float f) {
    union { float f; unsigned u; } c;
    c.f = f;
    unsigned u = c.u;
    // round-to-nearest-even truncation to bf16
    return (unsigned short)((u + 0x7FFFu + ((u >> 16) & 1u)) >> 16);
}

// ---------------------------------------------------------------------------
// Kernel 1: LayerNorm, one wave (32 lanes) per 256-wide row, 8 rows per block.
// Writes normalized x (f32) into Xn (aliased onto d_out as scratch).
// ---------------------------------------------------------------------------
__global__ __launch_bounds__(256) void ln_kernel(
    const float* __restrict__ x, const float* __restrict__ gamma,
    const float* __restrict__ beta, float* __restrict__ y) {
    int row  = blockIdx.x * 8 + (threadIdx.x >> 5);
    int lane = threadIdx.x & 31;

    const float4* px = (const float4*)(x + (size_t)row * C_) + lane * 2;
    float4 a = px[0];
    float4 b = px[1];

    float s = a.x + a.y + a.z + a.w + b.x + b.y + b.z + b.w;
    float q = a.x*a.x + a.y*a.y + a.z*a.z + a.w*a.w
            + b.x*b.x + b.y*b.y + b.z*b.z + b.w*b.w;
    #pragma unroll
    for (int m = 16; m >= 1; m >>= 1) {
        s += __shfl_xor(s, m, 32);
        q += __shfl_xor(q, m, 32);
    }
    float mu  = s * (1.0f / C_);
    float var = q * (1.0f / C_) - mu * mu;
    float inv = rsqrtf(var + 1e-5f);

    float4 g0 = ((const float4*)gamma)[lane * 2];
    float4 g1 = ((const float4*)gamma)[lane * 2 + 1];
    float4 t0 = ((const float4*)beta)[lane * 2];
    float4 t1 = ((const float4*)beta)[lane * 2 + 1];

    float4 o0, o1;
    o0.x = (a.x - mu) * inv * g0.x + t0.x;
    o0.y = (a.y - mu) * inv * g0.y + t0.y;
    o0.z = (a.z - mu) * inv * g0.z + t0.z;
    o0.w = (a.w - mu) * inv * g0.w + t0.w;
    o1.x = (b.x - mu) * inv * g1.x + t1.x;
    o1.y = (b.y - mu) * inv * g1.y + t1.y;
    o1.z = (b.z - mu) * inv * g1.z + t1.z;
    o1.w = (b.w - mu) * inv * g1.w + t1.w;

    float4* py = (float4*)(y + (size_t)row * C_) + lane * 2;
    py[0] = o0;
    py[1] = o1;
}

// ---------------------------------------------------------------------------
// Kernel 2: zero the aggregation buffers (grid-stride, float4 stores).
// ---------------------------------------------------------------------------
__global__ void zero_kernel(float4* __restrict__ p, int n4) {
    int i = blockIdx.x * blockDim.x + threadIdx.x;
    int stride = gridDim.x * blockDim.x;
    float4 z = make_float4(0.f, 0.f, 0.f, 0.f);
    for (; i < n4; i += stride) p[i] = z;
}

// ---------------------------------------------------------------------------
// Kernel 3: transpose + convert W [256,768] f32 -> Wt [768][256] bf16 so that
// B-fragments are contiguous 16B loads per lane.
// ---------------------------------------------------------------------------
__global__ __launch_bounds__(256) void wconv_kernel(
    const float* __restrict__ W, unsigned short* __restrict__ Wt) {
    int col = blockIdx.x;     // 0..767
    int k   = threadIdx.x;    // 0..255
    Wt[(size_t)col * C_ + k] = f2bf(W[(size_t)k * G_ + col]);
}

// ---------------------------------------------------------------------------
// Kernel 4: edge aggregation on the NARROW (256-wide) features.
//   Xagg[b,dst,:] += ew * Xn[b,src,:]      (scatter commuted before the GEMM)
//   Hagg[b,dst,:] += ew * hidden[b,src,:]
// One block per (edge, batch); 256 threads = one channel each; f32 atomics.
// ---------------------------------------------------------------------------
__global__ __launch_bounds__(256) void edge_agg_kernel(
    const float* __restrict__ Xn, const float* __restrict__ Hdn,
    const int* __restrict__ src, const int* __restrict__ dst,
    const float* __restrict__ ew,
    float* __restrict__ Xagg, float* __restrict__ Hagg) {
    int e = blockIdx.x;
    int b = blockIdx.y;
    int c = threadIdx.x;
    int s = src[e];
    int d = dst[e];
    float w = ew[e];
    size_t so   = ((size_t)b * N_ + s) * C_ + c;
    size_t dofs = ((size_t)b * N_ + d) * C_ + c;
    atomicAdd(Xagg + dofs, w * Xn[so]);
    atomicAdd(Hagg + dofs, w * Hdn[so]);
}

// ---------------------------------------------------------------------------
// Kernel 5: narrow the f32 aggregation results to bf16 in one elementwise
// pass (hoists all conversions out of the GEMM hot loop). 4 elems / thread.
// ---------------------------------------------------------------------------
__global__ __launch_bounds__(256) void bfpack_kernel(
    const float* __restrict__ s, unsigned short* __restrict__ d, int n) {
    int i = (blockIdx.x * blockDim.x + threadIdx.x) * 4;
    if (i >= n) return;
    float4 v = *(const float4*)(s + i);
    ushort4 o;
    o.x = f2bf(v.x);
    o.y = f2bf(v.y);
    o.z = f2bf(v.z);
    o.w = f2bf(v.w);
    *(ushort4*)(d + i) = o;
}

// ---------------------------------------------------------------------------
// Kernel 6: fused bf16 WMMA GEMM + GRU gates.
// Each wave owns a 16x16 output tile: 6 accumulators (r/i/n  x  {x,h} paths),
// K=256 in 8 steps of v_wmma_f32_16x16x32_bf16 (48 WMMAs / tile). All A and B
// fragments are pure b128 loads (bf16 pre-packed); then sigmoid/tanh gate
// math and the final output write — gi/gh never touch memory.
// ---------------------------------------------------------------------------
__global__ __launch_bounds__(256) void gemm_gru_kernel(
    const unsigned short* __restrict__ Xb, const unsigned short* __restrict__ Hb,
    const unsigned short* __restrict__ Wti, const unsigned short* __restrict__ Wth,
    const float* __restrict__ b_i, const float* __restrict__ b_h,
    const float* __restrict__ hidden, float* __restrict__ out) {
    int wave = blockIdx.x * 8 + (threadIdx.x >> 5);  // 0..65535
    int lane = threadIdx.x & 31;
    int tm = wave >> 4;          // 0..4095 row tile
    int tn = wave & 15;          // 0..15   col tile within 256
    int row0 = tm * 16;
    int col0 = tn * 16;

    // A layout (16x32 bf16): lane = M (mod 16); lanes<16 K=0-7,16-23; lanes>=16 K=8-15,24-31
    int mrow = row0 + (lane & 15);
    int kbA  = (lane >> 4) * 8;
    // B layout (32x16 bf16): lane = N (mod 16); lanes<16 K=0-7,16-23; lanes>=16 K=8-15,24-31
    int ncol = col0 + (lane & 15);
    int kbB  = (lane >> 4) * 8;

    const unsigned short* pax = Xb + (size_t)mrow * C_ + kbA;
    const unsigned short* pah = Hb + (size_t)mrow * C_ + kbA;

    v8f accI[3] = {v8f{}, v8f{}, v8f{}};
    v8f accH[3] = {v8f{}, v8f{}, v8f{}};

    #pragma unroll 2
    for (int kk = 0; kk < C_; kk += 32) {
        // --- A fragments: two 16B loads each (halves 0-7: K=kb..kb+7,
        //     halves 8-15: K=kb+16..kb+23) ---
        FragBF ax, ah;
        const uint4* qa = (const uint4*)(pax + kk);
        const uint4* qh = (const uint4*)(pah + kk);
        *(uint4*)&ax.u[0] = qa[0];
        *(uint4*)&ax.u[8] = qa[2];
        *(uint4*)&ah.u[0] = qh[0];
        *(uint4*)&ah.u[8] = qh[2];

        // --- B fragments: 3 gate column blocks per weight matrix ---
        #pragma unroll
        for (int g = 0; g < 3; ++g) {
            FragBF bi, bh;
            const uint4* qi = (const uint4*)(Wti + (size_t)(g * 256 + ncol) * C_ + kk + kbB);
            const uint4* qw = (const uint4*)(Wth + (size_t)(g * 256 + ncol) * C_ + kk + kbB);
            *(uint4*)&bi.u[0] = qi[0];
            *(uint4*)&bi.u[8] = qi[2];
            *(uint4*)&bh.u[0] = qw[0];
            *(uint4*)&bh.u[8] = qw[2];

            accI[g] = __builtin_amdgcn_wmma_f32_16x16x32_bf16(
                false, ax.v, false, bi.v, (short)0, accI[g], false, false);
            accH[g] = __builtin_amdgcn_wmma_f32_16x16x32_bf16(
                false, ah.v, false, bh.v, (short)0, accH[g], false, false);
        }
    }

    // --- GRU gate epilogue ---
    // C/D layout: VGPR r, lanes 0-15 -> M=r, lanes 16-31 -> M=r+8; N = lane&15
    int gcol = col0 + (lane & 15);
    float bir = b_i[gcol],       bhr = b_h[gcol];
    float bii = b_i[256 + gcol], bhi = b_h[256 + gcol];
    float bin = b_i[512 + gcol], bhn = b_h[512 + gcol];

    #pragma unroll
    for (int r = 0; r < 8; ++r) {
        int grow = row0 + r + (lane >> 4) * 8;
        float ir  = accI[0][r] + bir;
        float ii  = accI[1][r] + bii;
        float inn = accI[2][r] + bin;
        float hr  = accH[0][r] + bhr;
        float hi  = accH[1][r] + bhi;
        float hn  = accH[2][r] + bhn;
        float rg = 1.0f / (1.0f + __expf(-(ir + hr)));
        float ig = 1.0f / (1.0f + __expf(-(ii + hi)));
        float ng = tanhf(inn + rg * hn);
        float h  = hidden[(size_t)grow * C_ + gcol];
        out[(size_t)grow * C_ + gcol] = ng + ig * (h - ng);
    }
}

// ---------------------------------------------------------------------------
extern "C" void kernel_launch(void* const* d_in, const int* in_sizes, int n_in,
                              void* d_out, int out_size, void* d_ws, size_t ws_size,
                              hipStream_t stream) {
    const float* input  = (const float*)d_in[0];
    const float* hidden = (const float*)d_in[1];
    const float* gamma  = (const float*)d_in[2];
    const float* beta   = (const float*)d_in[3];
    const float* W_i    = (const float*)d_in[4];
    const float* b_i    = (const float*)d_in[5];
    const float* W_h    = (const float*)d_in[6];
    const float* b_h    = (const float*)d_in[7];
    const int*   esrc   = (const int*)d_in[8];
    const int*   edst   = (const int*)d_in[9];
    const float* ew     = (const float*)d_in[10];

    float* out = (float*)d_out;

    // Workspace layout:
    //   phase 1: [Xagg f32 64MB][Hagg f32 64MB][Wt_i bf16][Wt_h bf16]
    //   phase 2: [Xb bf16 32MB][Hb bf16 32MB][ dead f32  ][Wt_i][Wt_h]
    float* Xagg = (float*)d_ws;
    float* Hagg = Xagg + (size_t)M_ * C_;
    unsigned short* Wti = (unsigned short*)(Hagg + (size_t)M_ * C_);
    unsigned short* Wth = Wti + (size_t)G_ * C_;

    // Xn (LayerNorm output) temporarily lives in d_out; fully consumed by
    // edge_agg_kernel. Then d_out is reused as the bf16 staging area before
    // gemm_gru_kernel finally overwrites it with the real output.
    float* Xn = out;

    // 1) LayerNorm
    ln_kernel<<<M_ / 8, 256, 0, stream>>>(input, gamma, beta, Xn);

    // 2) Zero Xagg + Hagg (contiguous region)
    int n4 = 2 * M_ * C_ / 4;
    zero_kernel<<<4096, 256, 0, stream>>>((float4*)Xagg, n4);

    // 3) Weight transpose + bf16 convert
    wconv_kernel<<<G_, 256, 0, stream>>>(W_i, Wti);
    wconv_kernel<<<G_, 256, 0, stream>>>(W_h, Wth);

    // 4) Edge aggregation (narrow features, f32 atomics through L2)
    edge_agg_kernel<<<dim3(E_, B_), 256, 0, stream>>>(Xn, hidden, esrc, edst, ew,
                                                      Xagg, Hagg);

    // 5) Pack Xagg|Hagg (contiguous 128MB f32) -> bf16 staged in d_out (64MB)
    int nconv = 2 * M_ * C_;
    bfpack_kernel<<<nconv / (256 * 4), 256, 0, stream>>>(
        Xagg, (unsigned short*)out, nconv);

    // 6) Move packed bf16 back over the (now dead) f32 aggregation region
    hipMemcpyAsync(d_ws, d_out, (size_t)nconv * sizeof(unsigned short),
                   hipMemcpyDeviceToDevice, stream);

    const unsigned short* Xb = (const unsigned short*)d_ws;
    const unsigned short* Hb = Xb + (size_t)M_ * C_;

    // 7) Fused WMMA GEMM + GRU gates
    gemm_gru_kernel<<<(M_ * 16) / (16 * 8), 256, 0, stream>>>(
        Xb, Hb, Wti, Wth, b_i, b_h, hidden, out);
}
